// CMRLayer_47777216201163
// MI455X (gfx1250) — matrix operand order repaired and checked by reading, not compiled, not run on hardware
//
#include <hip/hip_runtime.h>

typedef __attribute__((ext_vector_type(16))) _Float16 v16h;
typedef __attribute__((ext_vector_type(8)))  float    v8f;

#define BB 2048
#define SS 8
#define DD 1280
#define PITCH 1282          // f16 row pitch: 641 words (odd) -> conflict-free columns
#define HIGH_T 0.6f
#define STORM_T 0.8f

__global__ __launch_bounds__(256)
void cmr_kernel(const float* __restrict__ feat,
                const float* __restrict__ kp,
                const float* __restrict__ cmr_gate,
                float* __restrict__ out)
{
    __shared__ _Float16 fnS[16 * PITCH];   // standardized features, f16 (two batches)
    __shared__ float gram[256];            // 16x16 (two 8x8 Gram blocks on diagonal)
    __shared__ float sigS[16], uS[16], coefS[16];
    __shared__ float kgS[2];
    __shared__ float gateS;

    const int tid  = threadIdx.x;
    const int lane = tid & 31;             // wave32 on gfx1250
    const int w    = tid >> 5;             // 8 waves
    const int blk  = blockIdx.x;           // handles batches 2*blk, 2*blk+1

    gram[tid] = 0.f;

    // ---------------- Pass 1: per-row mean/std + f16 normalized rows to LDS ----
    for (int rr = 0; rr < 2; ++rr) {
        const int r = 2 * w + rr;                       // row 0..15 (= batch pair x 8)
        const long base = (long)(16 * blk + r) * DD;
        float4 x[10];
        float sum = 0.f, sq = 0.f;
        #pragma unroll
        for (int i = 0; i < 10; ++i) {
            x[i] = *(const float4*)(feat + base + 4 * lane + 128 * i);
            sum += x[i].x + x[i].y + x[i].z + x[i].w;
            sq  += x[i].x * x[i].x + x[i].y * x[i].y + x[i].z * x[i].z + x[i].w * x[i].w;
        }
        #pragma unroll
        for (int m = 16; m >= 1; m >>= 1) {
            sum += __shfl_xor(sum, m, 32);
            sq  += __shfl_xor(sq,  m, 32);
        }
        const float mean = sum * (1.f / DD);
        const float var  = fmaxf(sq * (1.f / DD) - mean * mean, 0.f);
        const float sig  = sqrtf(var) + 1e-8f;          // numpy std + 1e-8
        const float rsig = 1.f / sig;
        _Float16* dst = fnS + r * PITCH + 4 * lane;
        #pragma unroll
        for (int i = 0; i < 10; ++i) {
            dst[128 * i + 0] = (_Float16)((x[i].x - mean) * rsig);
            dst[128 * i + 1] = (_Float16)((x[i].y - mean) * rsig);
            dst[128 * i + 2] = (_Float16)((x[i].z - mean) * rsig);
            dst[128 * i + 3] = (_Float16)((x[i].w - mean) * rsig);
        }
        if (lane == 0) sigS[r] = sig;
    }
    __syncthreads();

    // ---------------- Pass 2: Gram = fn * fn^T via WMMA (K split over 8 waves) --
    {
        const int half = lane >> 4;        // K-half selector per ISA fragment layout
        const int mrow = lane & 15;        // A: M row; B: N col; C: N col
        v8f c = {};
        #pragma unroll
        for (int i = 0; i < 5; ++i) {
            const int kb = w * 160 + 32 * i;
            v16h a, b;
            // A 16x32 f16: VGPR v<4 -> K = 2v + 8*half; v>=4 -> K = 16 + 2(v-4) + 8*half
            #pragma unroll
            for (int v = 0; v < 4; ++v) {
                const int k = kb + 2 * v + 8 * half;
                a[2 * v]     = fnS[mrow * PITCH + k];
                a[2 * v + 1] = fnS[mrow * PITCH + k + 1];
            }
            #pragma unroll
            for (int v = 4; v < 8; ++v) {
                const int k = kb + 16 + 2 * (v - 4) + 8 * half;
                a[2 * v]     = fnS[mrow * PITCH + k];
                a[2 * v + 1] = fnS[mrow * PITCH + k + 1];
            }
            // B 32x16 f16: VGPR v -> K = 2v + 8*half (+8 for v>=4), same as A ; B(k,n) = fn[n][k]
            #pragma unroll
            for (int v = 0; v < 8; ++v) {
                const int k = kb + 2 * v + 8 * half + (v >= 4 ? 8 : 0);
                b[2 * v]     = fnS[mrow * PITCH + k];
                b[2 * v + 1] = fnS[mrow * PITCH + k + 1];
            }
            c = __builtin_amdgcn_wmma_f32_16x16x32_f16(false, a, false, b,
                                                       (short)0, c, false, false);
        }
        // C layout: VGPR v holds M = v + 8*half, N = lane&15 -> ds_add_f32 reduce
        #pragma unroll
        for (int v = 0; v < 8; ++v)
            atomicAdd(&gram[(v + 8 * half) * 16 + mrow], c[v]);
    }
    __syncthreads();

    // ---------------- Pass 3: top eigenpair of each 8x8 Gram (power iteration) --
    if (tid < 2) {
        float G[8][8];
        float trace = 0.f;
        for (int i = 0; i < 8; ++i)
            for (int j = 0; j < 8; ++j)
                G[i][j] = gram[(8 * tid + i) * 16 + (8 * tid + j)];
        for (int i = 0; i < 8; ++i) trace += G[i][i];

        float v[8];
        for (int i = 0; i < 8; ++i) v[i] = 1.f;
        for (int it = 0; it < 40; ++it) {
            float wv[8]; float n2 = 0.f;
            for (int i = 0; i < 8; ++i) {
                float a = 0.f;
                for (int j = 0; j < 8; ++j) a += G[i][j] * v[j];
                wv[i] = a; n2 += a * a;
            }
            const float inv = rsqrtf(fmaxf(n2, 1e-30f));
            for (int i = 0; i < 8; ++i) v[i] = wv[i] * inv;
        }
        float lam = 0.f;
        for (int i = 0; i < 8; ++i) {
            float a = 0.f;
            for (int j = 0; j < 8; ++j) a += G[i][j] * v[j];
            lam += v[i] * a;                           // Rayleigh quotient = sv0^2
        }
        const float pc1 = lam / (trace + 1e-8f);
        const int flag = (pc1 >= HIGH_T ? 1 : 0) + (pc1 >= STORM_T ? 1 : 0);
        const int batch = 2 * blk + tid;
        out[(long)BB * SS * DD + batch]      = pc1;
        out[(long)BB * SS * DD + BB + batch] = (float)flag;
        for (int i = 0; i < 8; ++i) uS[8 * tid + i] = v[i];
        kgS[tid] = (kp[batch] >= 5.f) ? 1.5f : 1.f;    // kp gating
    }
    if (tid == 0) gateS = 1.f / (1.f + expf(-cmr_gate[0]));  // sigmoid gate
    __syncthreads();

    // out = features - gate*kp_gate*sigma_s*u_s*p_d  (clean-features == -low*sigma)
    if (tid < 16) coefS[tid] = gateS * kgS[tid >> 3] * sigS[tid] * uS[tid];
    __syncthreads();

    // ---------------- Pass 4: rank-1 apply, single coalesced stream out ---------
    #pragma unroll
    for (int bi = 0; bi < 2; ++bi) {
        #pragma unroll
        for (int j = 0; j < 5; ++j) {
            const int d = 256 * j + tid;
            float p = 0.f;
            #pragma unroll
            for (int t = 0; t < 8; ++t)
                p += (float)fnS[(8 * bi + t) * PITCH + d] * uS[8 * bi + t];
            #pragma unroll
            for (int s = 0; s < 8; ++s) {
                const int r = 8 * bi + s;
                const long idx = (long)(16 * blk + r) * DD + d;
                out[idx] = feat[idx] - coefS[r] * p;   // feat re-read hits 192MB L2
            }
        }
    }
}

extern "C" void kernel_launch(void* const* d_in, const int* in_sizes, int n_in,
                              void* d_out, int out_size, void* d_ws, size_t ws_size,
                              hipStream_t stream) {
    (void)in_sizes; (void)n_in; (void)d_ws; (void)ws_size; (void)out_size;
    const float* feat = (const float*)d_in[0];
    const float* kp   = (const float*)d_in[1];
    const float* gate = (const float*)d_in[2];
    float* out = (float*)d_out;
    cmr_kernel<<<BB / 2, 256, 0, stream>>>(feat, kp, gate, out);
}
